// CSOCSSC_v50_2319282340047
// MI455X (gfx1250) — matrix operand order repaired and checked by reading, not compiled, not run on hardware
//
#include <hip/hip_runtime.h>
#include <hip/hip_bf16.h>
#include <math.h>

typedef __attribute__((ext_vector_type(16))) _Float16 v16h;
typedef __attribute__((ext_vector_type(8)))  _Float16 v8h;
typedef __attribute__((ext_vector_type(8)))  float    v8f;

#define N_NODES   50000
#define N_EDGES   800000
#define NODE_DIM  64
#define EDGE_DIM  32
#define HID       128
#define MIN_DIM   160           // 2*NODE_DIM + EDGE_DIM
#define EDGES_PER_WAVE 16
#define WAVES_PER_BLOCK 8
#define EDGES_PER_BLOCK (EDGES_PER_WAVE * WAVES_PER_BLOCK)

// ws layout (_Float16 elements)
#define WS_WN1_OFF 0            // 8 ntiles * 5 kchunks * 32 lanes * 16 halves = 20480
#define WS_WC1_OFF 20480        // same size
#define WS_WN2_OFF 40960        // 4 * 4 * 32 * 16 = 8192
#define WS_WE2_OFF 49152        // 2 * 1 * 32 * 16 = 1024

__device__ __forceinline__ float silu_f(float v) {
    return v / (1.0f + __expf(-v));
}

__device__ __forceinline__ v16h cat16(v8h lo, v8h hi) {
    v16h r;
#pragma unroll
    for (int i = 0; i < 8; ++i) { r[i] = lo[i]; r[i + 8] = hi[i]; }
    return r;
}

// Repack a row-major [K, N] f32 weight into f16 WMMA B-fragments:
// out[(((nt*nK32)+kc)*32 + lane)*16 + idx] = W[(kc*32 + (lane>>4)*8 + (idx&7) + (idx>>3)*16)*ldN + nt*16 + (lane&15)]
__global__ void egnn_pack_bfrag(const float* __restrict__ W, _Float16* __restrict__ out,
                                int nN16, int nK32, int ldN) {
    int t = blockIdx.x * blockDim.x + threadIdx.x;
    int total = nN16 * nK32 * 32 * 16;
    if (t >= total) return;
    int idx  = t & 15;
    int lane = (t >> 4) & 31;
    int kc   = (t >> 9) % nK32;
    int nt   = (t >> 9) / nK32;
    int k = kc * 32 + (lane >> 4) * 8 + (idx & 7) + (idx >> 3) * 16;
    int n = nt * 16 + (lane & 15);
    out[t] = (_Float16)W[k * ldN + n];
}

__global__ void egnn_init_out(const float* __restrict__ h, const float* __restrict__ x,
                              float* __restrict__ out) {
    int t = blockIdx.x * blockDim.x + threadIdx.x;
    const int NH = N_NODES * NODE_DIM;
    const int NX = N_NODES * 3;
    if (t < NH)           out[t] = h[t];
    else if (t < NH + NX) out[t] = x[t - NH];
}

__global__ __launch_bounds__(256) void egnn_edge_kernel(
    const float* __restrict__ h, const float* __restrict__ x,
    const float* __restrict__ edge_dist,
    const float* __restrict__ W_e1, const float* __restrict__ b_e1,
    const float* __restrict__ b_e2,
    const float* __restrict__ b_n1, const float* __restrict__ b_n2,
    const float* __restrict__ b_c1, const float* __restrict__ W_c2,
    const int* __restrict__ edge_idx,
    const _Float16* __restrict__ wsWn1, const _Float16* __restrict__ wsWc1,
    const _Float16* __restrict__ wsWn2, const _Float16* __restrict__ wsWe2,
    float* __restrict__ out_h, float* __restrict__ out_x)
{
    // per-wave m_in tile [16 edges][160]; hid tile aliases cols 0..127 after
    // A-fragments are resident in VGPRs; edge-attr cols 128..159 are first the
    // SiLU outputs t, then (post edge-WMMA) the edge_attr values.
    __shared__ __align__(16) _Float16 smem[WAVES_PER_BLOCK][EDGES_PER_WAVE][MIN_DIM];

    const int wave = threadIdx.x >> 5;
    const int lane = threadIdx.x & 31;
    const int m    = lane & 15;    // row within 16-wide tile (M index / edge)
    const int hi   = lane >> 4;    // half-wave selector
    const long tileBase = ((long)blockIdx.x * WAVES_PER_BLOCK + wave) * EDGES_PER_WAVE;

    // ---------- Phase 1: gather h and drop edge-MLP layer-1 activations ----------
    {
        const int e    = m;        // edge slot handled by this lane
        const int part = hi;       // 0: h[src] + t[0..15], 1: h[dst] + t[16..31]
        long eg = tileBase + e;
        long egc = eg < N_EDGES ? eg : (N_EDGES - 1);
        int srcn = edge_idx[egc];
        int dstn = edge_idx[N_EDGES + egc];
        float d  = (eg < N_EDGES) ? edge_dist[egc] : 0.0f;

        // edge MLP layer 1: t_j = silu(d*W_e1[j] + b_e1[j]); each half does 16 j's
#pragma unroll
        for (int j0 = 0; j0 < 16; ++j0) {
            int j = part * 16 + j0;
            smem[wave][e][2 * NODE_DIM + j] = (_Float16)silu_f(d * W_e1[j] + b_e1[j]);
        }

        // gather h[src] (part 0 -> cols 0..63) / h[dst] (part 1 -> cols 64..127)
        const float4* hp4 = (const float4*)(h + (size_t)(part ? dstn : srcn) * NODE_DIM);
        int cbase = part * NODE_DIM;
#pragma unroll
        for (int i = 0; i < NODE_DIM / 4; ++i) {
            float4 v = hp4[i];
            int c = cbase + i * 4;
            smem[wave][e][c + 0] = (_Float16)v.x;
            smem[wave][e][c + 1] = (_Float16)v.y;
            smem[wave][e][c + 2] = (_Float16)v.z;
            smem[wave][e][c + 3] = (_Float16)v.w;
        }
    }
    __syncthreads();

    // ---------- Phase 1b: edge MLP layer 2 as WMMA: t[16x32] @ W_e2[32x32] + b_e2 ----
    {
        // A-fragment over the t values (cols 128..159)
        const v8h* q0 = (const v8h*)&smem[wave][m][2 * NODE_DIM + hi * 8];
        const v8h* q1 = (const v8h*)&smem[wave][m][2 * NODE_DIM + hi * 8 + 16];
        v16h at = cat16(*q0, *q1);
#pragma unroll
        for (int nt = 0; nt < 2; ++nt) {
            float bias = b_e2[nt * 16 + m];
            v8f acc;
#pragma unroll
            for (int r = 0; r < 8; ++r) acc[r] = bias;
            const v8h* bp = (const v8h*)(wsWe2 + (size_t)(nt * 32 + lane) * 16);
            v16h bfr = cat16(bp[0], bp[1]);
            acc = __builtin_amdgcn_wmma_f32_16x16x32_f16(
                false, at, false, bfr, (short)0, acc, false, false);
            // D layout: VGPR r holds row M = r + hi*8, col N = lane%16
#pragma unroll
            for (int r = 0; r < 8; ++r) {
                int row = r + hi * 8;
                smem[wave][row][2 * NODE_DIM + nt * 16 + m] = (_Float16)acc[r];
            }
        }
    }
    __syncthreads();

    // ---------- Phase 2: A-fragments (16x32 f16) for all 5 K-chunks ----------
    // ISA A layout: lane L holds row M=L%16, K = (L/16)*8 + {0..7} in halves 0..7
    // and K = (L/16)*8 + 16 + {0..7} in halves 8..15.
    v16h afr[5];
#pragma unroll
    for (int kc = 0; kc < 5; ++kc) {
        const v8h* p0 = (const v8h*)&smem[wave][m][kc * 32 + hi * 8];
        const v8h* p1 = (const v8h*)&smem[wave][m][kc * 32 + hi * 8 + 16];
        afr[kc] = cat16(*p0, *p1);
    }
    __syncthreads();

    // ---------- Phase 3: node hidden = silu(m_in @ W_n1 + b_n1) -> LDS ----------
#pragma unroll
    for (int nt = 0; nt < 8; ++nt) {
        float bias = b_n1[nt * 16 + m];
        v8f acc;
#pragma unroll
        for (int r = 0; r < 8; ++r) acc[r] = bias;
#pragma unroll
        for (int kc = 0; kc < 5; ++kc) {
            const v8h* bp = (const v8h*)(wsWn1 + (size_t)(((nt * 5) + kc) * 32 + lane) * 16);
            v16h bfr = cat16(bp[0], bp[1]);
            acc = __builtin_amdgcn_wmma_f32_16x16x32_f16(
                false, afr[kc], false, bfr, (short)0, acc, false, false);
        }
#pragma unroll
        for (int r = 0; r < 8; ++r) {
            int row = r + hi * 8;
            smem[wave][row][nt * 16 + m] = (_Float16)silu_f(acc[r]);
        }
    }
    __syncthreads();

    // ---------- Phase 4: m = hid @ W_n2 + b_n2, atomic scatter into out_h ----------
    {
        v16h a2[4];
#pragma unroll
        for (int kc = 0; kc < 4; ++kc) {
            const v8h* p0 = (const v8h*)&smem[wave][m][kc * 32 + hi * 8];
            const v8h* p1 = (const v8h*)&smem[wave][m][kc * 32 + hi * 8 + 16];
            a2[kc] = cat16(*p0, *p1);
        }
#pragma unroll
        for (int nt2 = 0; nt2 < 4; ++nt2) {
            float bias = b_n2[nt2 * 16 + m];
            v8f acc;
#pragma unroll
            for (int r = 0; r < 8; ++r) acc[r] = bias;
#pragma unroll
            for (int kc = 0; kc < 4; ++kc) {
                const v8h* bp = (const v8h*)(wsWn2 + (size_t)(((nt2 * 4) + kc) * 32 + lane) * 16);
                v16h bfr = cat16(bp[0], bp[1]);
                acc = __builtin_amdgcn_wmma_f32_16x16x32_f16(
                    false, a2[kc], false, bfr, (short)0, acc, false, false);
            }
#pragma unroll
            for (int r = 0; r < 8; ++r) {
                int row = r + hi * 8;
                long eg2 = tileBase + row;
                if (eg2 < N_EDGES) {
                    int d2 = edge_idx[N_EDGES + eg2];
                    atomicAdd(&out_h[(size_t)d2 * NODE_DIM + nt2 * 16 + m], acc[r]);
                }
            }
        }
    }
    __syncthreads();

    // ---------- Phase 5: coord hidden = silu(m_in @ W_c1 + b_c1) -> LDS ----------
#pragma unroll
    for (int nt = 0; nt < 8; ++nt) {
        float bias = b_c1[nt * 16 + m];
        v8f acc;
#pragma unroll
        for (int r = 0; r < 8; ++r) acc[r] = bias;
#pragma unroll
        for (int kc = 0; kc < 5; ++kc) {
            const v8h* bp = (const v8h*)(wsWc1 + (size_t)(((nt * 5) + kc) * 32 + lane) * 16);
            v16h bfr = cat16(bp[0], bp[1]);
            acc = __builtin_amdgcn_wmma_f32_16x16x32_f16(
                false, afr[kc], false, bfr, (short)0, acc, false, false);
        }
#pragma unroll
        for (int r = 0; r < 8; ++r) {
            int row = r + hi * 8;
            smem[wave][row][nt * 16 + m] = (_Float16)silu_f(acc[r]);
        }
    }
    __syncthreads();

    // ---------- Phase 6: coord_w = hid @ W_c2 (N=1, VALU dot), coord scatter ----------
    {
        float s = 0.0f;
#pragma unroll 16
        for (int k0 = 0; k0 < 64; ++k0) {
            int k = hi * 64 + k0;
            s += (float)smem[wave][m][k] * W_c2[k];
        }
        s += __shfl_xor(s, 16);    // combine the two K-halves; lanes 0..15 hold full dot

        long eg = tileBase + m;
        if (hi == 0 && eg < N_EDGES) {
            int srcn = edge_idx[eg];
            int dstn = edge_idx[N_EDGES + eg];
            float dx = x[srcn * 3 + 0] - x[dstn * 3 + 0];
            float dy = x[srcn * 3 + 1] - x[dstn * 3 + 1];
            float dz = x[srcn * 3 + 2] - x[dstn * 3 + 2];
            float len = fmaxf(sqrtf(dx * dx + dy * dy + dz * dz), 1e-8f);
            float w = s / len;
            atomicAdd(&out_x[dstn * 3 + 0], w * dx);
            atomicAdd(&out_x[dstn * 3 + 1], w * dy);
            atomicAdd(&out_x[dstn * 3 + 2], w * dz);
        }
    }
}

extern "C" void kernel_launch(void* const* d_in, const int* in_sizes, int n_in,
                              void* d_out, int out_size, void* d_ws, size_t ws_size,
                              hipStream_t stream) {
    const float* h         = (const float*)d_in[0];
    const float* x         = (const float*)d_in[1];
    const float* edge_dist = (const float*)d_in[2];
    const float* W_e1      = (const float*)d_in[3];
    const float* b_e1      = (const float*)d_in[4];
    const float* W_e2      = (const float*)d_in[5];
    const float* b_e2      = (const float*)d_in[6];
    const float* W_n1      = (const float*)d_in[7];
    const float* b_n1      = (const float*)d_in[8];
    const float* W_n2      = (const float*)d_in[9];
    const float* b_n2      = (const float*)d_in[10];
    const float* W_c1      = (const float*)d_in[11];
    const float* b_c1      = (const float*)d_in[12];
    const float* W_c2      = (const float*)d_in[13];
    const int*   edge_idx  = (const int*)d_in[14];

    float* out   = (float*)d_out;
    float* out_h = out;
    float* out_x = out + (size_t)N_NODES * NODE_DIM;

    _Float16* ws    = (_Float16*)d_ws;
    _Float16* wsWn1 = ws + WS_WN1_OFF;
    _Float16* wsWc1 = ws + WS_WC1_OFF;
    _Float16* wsWn2 = ws + WS_WN2_OFF;
    _Float16* wsWe2 = ws + WS_WE2_OFF;

    // repack weights into f16 WMMA B-fragment order
    egnn_pack_bfrag<<<(20480 + 255) / 256, 256, 0, stream>>>(W_n1, wsWn1, 8, 5, HID);
    egnn_pack_bfrag<<<(20480 + 255) / 256, 256, 0, stream>>>(W_c1, wsWc1, 8, 5, HID);
    egnn_pack_bfrag<<<(8192  + 255) / 256, 256, 0, stream>>>(W_n2, wsWn2, 4, 4, NODE_DIM);
    egnn_pack_bfrag<<<(1024  + 255) / 256, 256, 0, stream>>>(W_e2, wsWe2, 2, 1, EDGE_DIM);

    // out = (h, x) so edge kernel can accumulate with atomics
    const int initN = N_NODES * NODE_DIM + N_NODES * 3;
    egnn_init_out<<<(initN + 255) / 256, 256, 0, stream>>>(h, x, out);

    const int nblk = (N_EDGES + EDGES_PER_BLOCK - 1) / EDGES_PER_BLOCK;  // 6250
    egnn_edge_kernel<<<nblk, 256, 0, stream>>>(
        h, x, edge_dist, W_e1, b_e1, b_e2,
        b_n1, b_n2, b_c1, W_c2, edge_idx,
        wsWn1, wsWc1, wsWn2, wsWe2, out_h, out_x);
}
